// EnhancedGraphEncoder_61272003445439
// MI455X (gfx1250) — matrix operand order repaired and checked by reading, not compiled, not run on hardware
//
#include <hip/hip_runtime.h>
#include <math.h>

#define NNODE 50000
#define NPAD  50016             // NNODE padded to multiple of 32 (A-side reads only)
#define NEDGE 100000
#define NREL  8
#define HD    256
#define LD    768
#define OUTD  768
#define NG    64
#define EPADC 100352            // E + per-relation pad32 headroom, multiple of 32
#define NBLKE (EPADC/32)        // 3136 row-blocks of 32
#define GSCH  64                // node run length for pooling accumulation

typedef __attribute__((ext_vector_type(16))) __bf16 v16bf;
typedef __attribute__((ext_vector_type(8)))  float  v8f;

struct Q8 { uint4 lo, hi; };

__device__ __forceinline__ unsigned short f2bf(float f) {
  unsigned int u = __builtin_bit_cast(unsigned int, f);
  u += 0x7FFFu + ((u >> 16) & 1u);            // round-to-nearest-even
  return (unsigned short)(u >> 16);
}

__device__ __forceinline__ v16bf frag_pack(uint4 lo, uint4 hi) {
  Q8 q; q.lo = lo; q.hi = hi;
  return __builtin_bit_cast(v16bf, q);
}

// A fragment, 16x32 bf16, row-major A (lda=K).  ISA layout: lanes 0-15 M=lane,
// v0..3 hold K={0..7}+half*8, v4..7 K=+16.  Two b128 loads per lane.
__device__ __forceinline__ v16bf load_a_frag(const unsigned short* __restrict__ A,
                                             size_t row0, int K, int kb, int lane) {
  int half = lane >> 4, m = lane & 15;
  const unsigned short* p = A + (row0 + (size_t)m) * (size_t)K + (size_t)(kb + half * 8);
  return frag_pack(*(const uint4*)p, *(const uint4*)(p + 16));
}

// B fragment from pre-transposed weights Bt [Nout x K] row-major: lane holds
// column n = lane&15, k = {0..7}+half*8 and +16 (same order as A).
__device__ __forceinline__ v16bf load_b_frag(const unsigned short* __restrict__ Bt,
                                             int n0, int K, int kb, int lane) {
  int half = lane >> 4, n = lane & 15;
  const unsigned short* p = Bt + (size_t)(n0 + n) * (size_t)K + (size_t)(kb + half * 8);
  return frag_pack(*(const uint4*)p, *(const uint4*)(p + 16));
}

// ------ WMMA GEMM: C = A(bf16)[M,K] * Bt(bf16)[Nout,K]^T + bias -------------
// block = (32,4): 4 waves; wave computes a 32x64 tile (2 A-frags x 4 B-tiles,
// 8 WMMAs / K-step).  Rows padded to 32; stores guarded by M.
__global__ void __launch_bounds__(128)
k_gemm_wmma(const unsigned short* __restrict__ A, const unsigned short* __restrict__ Bt,
            float* __restrict__ Cf, unsigned short* __restrict__ Cb,
            const float* __restrict__ bias, int M, int K, int Nout, int relu)
{
  int lane = threadIdx.x;
  size_t row0 = (size_t)blockIdx.x * 32;
  int ncol0 = blockIdx.y * 256 + threadIdx.y * 64;
  v8f acc[4][2];
  #pragma unroll
  for (int t = 0; t < 4; t++)
    #pragma unroll
    for (int f = 0; f < 2; f++)
      #pragma unroll
      for (int j = 0; j < 8; j++) acc[t][f][j] = 0.0f;

  for (int kb = 0; kb < K; kb += 32) {
    v16bf a0 = load_a_frag(A, row0,      K, kb, lane);
    v16bf a1 = load_a_frag(A, row0 + 16, K, kb, lane);
    if (kb + 32 < K) {
      int half = lane >> 4, m = lane & 15;
      __builtin_prefetch(A + (row0 + (size_t)m) * (size_t)K + (size_t)(kb + 32 + half * 8), 0, 1);
      __builtin_prefetch(A + (row0 + 16 + (size_t)m) * (size_t)K + (size_t)(kb + 32 + half * 8), 0, 1);
    }
    #pragma unroll
    for (int t = 0; t < 4; t++) {
      v16bf b = load_b_frag(Bt, ncol0 + t * 16, K, kb, lane);
      acc[t][0] = __builtin_amdgcn_wmma_f32_16x16x32_bf16(false, a0, false, b,
                                                          (short)0, acc[t][0], false, false);
      acc[t][1] = __builtin_amdgcn_wmma_f32_16x16x32_bf16(false, a1, false, b,
                                                          (short)0, acc[t][1], false, false);
    }
  }
  // C/D layout: lane = n + 16*(m>=8), vgpr j -> m = 8*half + j
  int half = lane >> 4, n = lane & 15;
  #pragma unroll
  for (int t = 0; t < 4; t++) {
    int col = ncol0 + t * 16 + n;
    float bv = bias ? bias[col] : 0.0f;
    #pragma unroll
    for (int f = 0; f < 2; f++) {
      #pragma unroll
      for (int j = 0; j < 8; j++) {
        size_t r = row0 + (size_t)(f * 16 + half * 8 + j);
        if (r >= (size_t)M) continue;
        float v = acc[t][f][j] + bv;
        if (relu) v = fmaxf(v, 0.0f);
        if (Cf) Cf[r * (size_t)Nout + col] = v;
        if (Cb) Cb[r * (size_t)Nout + col] = f2bf(v);
      }
    }
  }
}

// -------- WMMA GEMM over gathered edge rows (32-row blocks, uniform relation
// per block); epilogue scatters with mean weight: out[dst] += val/cnt[r][dst] --
__global__ void __launch_bounds__(128)
k_gemm_edges(const unsigned short* __restrict__ As, const unsigned short* __restrict__ Wt,
             const int* __restrict__ blockrel, const int* __restrict__ sdst,
             const float* __restrict__ sw, float* __restrict__ outF, int K, int Nout)
{
  int lane = threadIdx.x;
  size_t row0 = (size_t)blockIdx.x * 32;
  int ncol0 = threadIdx.y * 64;
  const unsigned short* Bt = Wt + (size_t)blockrel[blockIdx.x] * (size_t)Nout * (size_t)K;
  v8f acc[4][2];
  #pragma unroll
  for (int t = 0; t < 4; t++)
    #pragma unroll
    for (int f = 0; f < 2; f++)
      #pragma unroll
      for (int j = 0; j < 8; j++) acc[t][f][j] = 0.0f;

  for (int kb = 0; kb < K; kb += 32) {
    v16bf a0 = load_a_frag(As, row0,      K, kb, lane);
    v16bf a1 = load_a_frag(As, row0 + 16, K, kb, lane);
    if (kb + 32 < K) {
      int half = lane >> 4, m = lane & 15;
      __builtin_prefetch(As + (row0 + (size_t)m) * (size_t)K + (size_t)(kb + 32 + half * 8), 0, 1);
      __builtin_prefetch(As + (row0 + 16 + (size_t)m) * (size_t)K + (size_t)(kb + 32 + half * 8), 0, 1);
    }
    #pragma unroll
    for (int t = 0; t < 4; t++) {
      v16bf b = load_b_frag(Bt, ncol0 + t * 16, K, kb, lane);
      acc[t][0] = __builtin_amdgcn_wmma_f32_16x16x32_bf16(false, a0, false, b,
                                                          (short)0, acc[t][0], false, false);
      acc[t][1] = __builtin_amdgcn_wmma_f32_16x16x32_bf16(false, a1, false, b,
                                                          (short)0, acc[t][1], false, false);
    }
  }
  int half = lane >> 4, n = lane & 15;
  #pragma unroll
  for (int f = 0; f < 2; f++) {
    #pragma unroll
    for (int j = 0; j < 8; j++) {
      size_t r = row0 + (size_t)(f * 16 + half * 8 + j);
      int d = sdst[r];
      if (d < 0) continue;                       // pad row
      float w = sw[r];
      size_t base = (size_t)d * (size_t)Nout;
      #pragma unroll
      for (int t = 0; t < 4; t++) {
        int col = ncol0 + t * 16 + n;
        unsafeAtomicAdd(&outF[base + col], acc[t][f][j] * w);
      }
    }
  }
}

// ---------------- small helper kernels ----------------
__global__ void k_zero_i32(int* p, long n) {
  long i = (long)blockIdx.x * blockDim.x + threadIdx.x; if (i < n) p[i] = 0;
}
__global__ void k_zero_f32(float* p, long n) {
  long i = (long)blockIdx.x * blockDim.x + threadIdx.x; if (i < n) p[i] = 0.0f;
}
__global__ void k_conv_bf16(const float* __restrict__ in, unsigned short* __restrict__ out, long n) {
  long i = (long)blockIdx.x * blockDim.x + threadIdx.x; if (i < n) out[i] = f2bf(in[i]);
}
// in [R][K][Nn] f32 -> out [R][Nn][K] bf16
__global__ void k_transpose_bf16(const float* __restrict__ in, unsigned short* __restrict__ out,
                                 int R_, int K_, int Nn) {
  long i = (long)blockIdx.x * blockDim.x + threadIdx.x;
  long tot = (long)R_ * K_ * Nn; if (i >= tot) return;
  int r = (int)(i / ((long)K_ * Nn));
  long rem = i - (long)r * K_ * Nn;
  int k = (int)(rem / Nn), n = (int)(rem % Nn);
  out[((size_t)r * Nn + n) * (size_t)K_ + k] = f2bf(in[i]);
}
__global__ void k_count(const int* __restrict__ src, const int* __restrict__ dst,
                        const int* __restrict__ attr, const int* __restrict__ batch,
                        int* cnt, int* rc, int* cgr) {
  int e = blockIdx.x * blockDim.x + threadIdx.x; if (e >= NEDGE) return;
  int r = attr[e];
  atomicAdd(&cnt[(size_t)r * NNODE + dst[e]], 1);
  atomicAdd(&rc[r], 1);
  atomicAdd(&cgr[batch[src[e]] * NREL + r], 1);
}
__global__ void k_offsets(const int* rc, int* poff, int* fc) {
  if (blockIdx.x == 0 && threadIdx.x == 0) {
    int o = 0;
    for (int r = 0; r < NREL; r++) { poff[r] = o; o += ((rc[r] + 31) / 32) * 32; fc[r] = 0; }
    poff[NREL] = o;
  }
}
__global__ void k_blockrel(const int* __restrict__ poff, int* __restrict__ blockrel) {
  int b = blockIdx.x * blockDim.x + threadIdx.x; if (b >= NBLKE) return;
  int row = b * 32, rel = 0;
  for (int r = 0; r < NREL; r++) if (row >= poff[r]) rel = r;
  blockrel[b] = rel;
}
__global__ void k_init_pad(int* ssrc, int* sdst, float* sw) {
  int i = blockIdx.x * blockDim.x + threadIdx.x; if (i >= EPADC) return;
  ssrc[i] = -1; sdst[i] = -1; sw[i] = 0.0f;
}
__global__ void k_build(const int* __restrict__ src, const int* __restrict__ dst,
                        const int* __restrict__ attr, const int* __restrict__ poff,
                        int* fc, const int* __restrict__ cnt,
                        int* ssrc, int* sdst, float* sw) {
  int e = blockIdx.x * blockDim.x + threadIdx.x; if (e >= NEDGE) return;
  int r = attr[e];
  int p = poff[r] + atomicAdd(&fc[r], 1);
  ssrc[p] = src[e]; sdst[p] = dst[e];
  int c = cnt[(size_t)r * NNODE + dst[e]];
  sw[p] = 1.0f / (float)(c > 0 ? c : 1);
}
__global__ void k_gather(const int* __restrict__ ssrc, const unsigned short* __restrict__ Hsrc,
                         unsigned short* __restrict__ As, int K) {
  int vecs = K / 8;
  long i = (long)blockIdx.x * blockDim.x + threadIdx.x;
  long tot = (long)EPADC * vecs; if (i >= tot) return;
  int p = (int)(i / vecs), t = (int)(i % vecs);
  uint4 v;
  int s = ssrc[p];
  if (s >= 0) v = *(const uint4*)(Hsrc + (size_t)s * K + (size_t)t * 8);
  else { v.x = 0; v.y = 0; v.z = 0; v.w = 0; }
  *(uint4*)(As + (size_t)p * K + (size_t)t * 8) = v;
}
__global__ void k_relu_bf(float* __restrict__ io, unsigned short* __restrict__ ob, long n) {
  long i = (long)blockIdx.x * blockDim.x + threadIdx.x; if (i >= n) return;
  float v = fmaxf(io[i], 0.0f); io[i] = v; ob[i] = f2bf(v);
}
__global__ void k_final_h(const float* __restrict__ h1, const float* __restrict__ o2,
                          float* __restrict__ out, long n) {
  long i = (long)blockIdx.x * blockDim.x + threadIdx.x; if (i >= n) return;
  out[i] = h1[i] + fmaxf(o2[i], 0.0f);
}
// batch is sorted: accumulate a run of GSCH nodes locally, flush on segment edge.
__global__ void k_gsum(const float* __restrict__ h, const int* __restrict__ batch,
                       float* __restrict__ gsum) {
  const int nchunks = (NNODE + GSCH - 1) / GSCH;
  long i = (long)blockIdx.x * blockDim.x + threadIdx.x;
  if (i >= (long)nchunks * HD) return;
  int chunk = (int)(i / HD), c = (int)(i % HD);
  int n0 = chunk * GSCH;
  int n1 = n0 + GSCH; if (n1 > NNODE) n1 = NNODE;
  float accum = 0.0f; int curb = batch[n0];
  for (int n = n0; n < n1; n++) {
    int b = batch[n];
    if (b != curb) { unsafeAtomicAdd(&gsum[curb * HD + c], accum); accum = 0.0f; curb = b; }
    accum += h[(size_t)n * HD + c];
  }
  unsafeAtomicAdd(&gsum[curb * HD + c], accum);
}
__global__ void k_gmean(const float* __restrict__ gsum, const int* __restrict__ gptr,
                        float* __restrict__ gmean) {
  int i = blockIdx.x * blockDim.x + threadIdx.x; if (i >= NG * HD) return;
  int g = i / HD;
  float c = (float)(gptr[g + 1] - gptr[g]);
  gmean[i] = gsum[i] / fmaxf(c, 1.0f);
}
// scalar f32 GEMM for tiny tails: C[m, n] (ldc) = act(A[m,K] @ B[K,Nn] + bias)
__global__ void k_mm_scalar(const float* __restrict__ A, const float* __restrict__ B,
                            const float* __restrict__ bias, float* __restrict__ C,
                            int M, int K, int Nn, int ldc, int act) {
  long i = (long)blockIdx.x * blockDim.x + threadIdx.x;
  if (i >= (long)M * Nn) return;
  int m = (int)(i / Nn), n = (int)(i % Nn);
  float s = bias ? bias[n] : 0.0f;
  const float* a = A + (size_t)m * K;
  for (int k = 0; k < K; k++) s = fmaf(a[k], B[(size_t)k * Nn + n], s);
  if (act == 1) s = tanhf(s);
  C[(size_t)m * ldc + n] = s;
}
__global__ void k_softmax_global(const float* __restrict__ sv, const int* __restrict__ rc,
                                 float* __restrict__ av) {
  if (blockIdx.x != 0 || threadIdx.x != 0) return;
  float mx = -1e30f;
  for (int r = 0; r < NREL; r++) if (rc[r] > 0) mx = fmaxf(mx, sv[r]);
  float den = 0.0f;
  for (int r = 0; r < NREL; r++) if (rc[r] > 0) den += (float)rc[r] * expf(sv[r] - mx);
  den = fmaxf(den, 1e-30f);
  for (int r = 0; r < NREL; r++) av[r] = (rc[r] > 0) ? expf(sv[r] - mx) / den : 0.0f;
}
__global__ void k_pw(const int* __restrict__ cgr, const float* __restrict__ av,
                     float* __restrict__ PW) {
  int g = blockIdx.x * blockDim.x + threadIdx.x; if (g >= NG) return;
  float gmax = -1e30f; int any = 0;
  for (int r = 0; r < NREL; r++) if (cgr[g * NREL + r] > 0) { any = 1; gmax = fmaxf(gmax, av[r]); }
  float den = 0.0f;
  for (int r = 0; r < NREL; r++) if (cgr[g * NREL + r] > 0)
    den += (float)cgr[g * NREL + r] * expf(av[r] - gmax);
  den = fmaxf(den, 1e-30f);
  for (int r = 0; r < NREL; r++)
    PW[g * NREL + r] = (any && cgr[g * NREL + r] > 0) ? expf(av[r] - gmax) / den : 0.0f;
}
// rel_ctx[g] = sum_r cnt[g,r] * per_edge_weight[g,r] * relproj[r]
__global__ void k_relctx(const float* __restrict__ PW, const int* __restrict__ cgr,
                         const float* __restrict__ relproj, float* __restrict__ combA) {
  long i = (long)blockIdx.x * blockDim.x + threadIdx.x;
  if (i >= (long)NG * LD) return;
  int g = (int)(i / LD), c = (int)(i % LD);
  float s = 0.0f;
  for (int r = 0; r < NREL; r++)
    s += (float)cgr[g * NREL + r] * PW[g * NREL + r] * relproj[(size_t)r * LD + c];
  combA[(size_t)g * (2 * LD) + LD + c] = s;
}
__global__ void __launch_bounds__(256)
k_layernorm(const float* __restrict__ comb, const float* __restrict__ gam,
            const float* __restrict__ bet, float* __restrict__ out) {
  __shared__ float red[256];
  int row = blockIdx.x, t = threadIdx.x;
  float s = 0.0f, q = 0.0f;
  for (int c = t; c < OUTD; c += 256) { float v = comb[(size_t)row * OUTD + c]; s += v; q += v * v; }
  red[t] = s; __syncthreads();
  for (int o = 128; o > 0; o >>= 1) { if (t < o) red[t] += red[t + o]; __syncthreads(); }
  float mean = red[0] / (float)OUTD; __syncthreads();
  red[t] = q; __syncthreads();
  for (int o = 128; o > 0; o >>= 1) { if (t < o) red[t] += red[t + o]; __syncthreads(); }
  float var = red[0] / (float)OUTD - mean * mean;
  float inv = rsqrtf(var + 1e-5f);
  for (int c = t; c < OUTD; c += 256) {
    float v = comb[(size_t)row * OUTD + c];
    out[(size_t)row * OUTD + c] = (v - mean) * inv * gam[c] + bet[c];
  }
}

static inline dim3 g1d(long n, int b = 256) { return dim3((unsigned)((n + b - 1) / b)); }

extern "C" void kernel_launch(void* const* d_in, const int* in_sizes, int n_in,
                              void* d_out, int out_size, void* d_ws, size_t ws_size,
                              hipStream_t stream) {
  (void)in_sizes; (void)n_in; (void)out_size; (void)ws_size;
  const float* x         = (const float*)d_in[0];
  const int*   eidx      = (const int*)d_in[1];
  const int*   eattr     = (const int*)d_in[2];
  const int*   batch     = (const int*)d_in[3];
  const int*   gptr      = (const int*)d_in[4];
  const float* W_init    = (const float*)d_in[5];
  const float* b_init    = (const float*)d_in[6];
  const float* rel_table = (const float*)d_in[7];
  const float* W_root1   = (const float*)d_in[8];
  const float* W_rel1    = (const float*)d_in[9];
  const float* b1        = (const float*)d_in[10];
  const float* W_root2   = (const float*)d_in[11];
  const float* W_rel2    = (const float*)d_in[12];
  const float* b2        = (const float*)d_in[13];
  const float* Wg        = (const float*)d_in[14];
  const float* bg        = (const float*)d_in[15];
  const float* Wa1       = (const float*)d_in[16];
  const float* ba1       = (const float*)d_in[17];
  const float* Wa2       = (const float*)d_in[18];
  const float* ba2       = (const float*)d_in[19];
  const float* Wo        = (const float*)d_in[20];
  const float* bo        = (const float*)d_in[21];
  const float* ln_g      = (const float*)d_in[22];
  const float* ln_b      = (const float*)d_in[23];

  const int* esrc = eidx;
  const int* edst = eidx + NEDGE;
  float* out_nodes = (float*)d_out;                           // [N,H]
  float* out_graph = (float*)d_out + (size_t)NNODE * HD;      // [G,OUT]

  char* p = (char*)d_ws;
  auto alloc = [&](size_t bytes) -> void* {
    void* r = (void*)p; p += (bytes + 255) & ~(size_t)255; return r;
  };
  unsigned short* hb    = (unsigned short*)alloc((size_t)NPAD * LD * 2);
  unsigned short* As    = (unsigned short*)alloc((size_t)EPADC * LD * 2);
  float*          out1F = (float*)alloc((size_t)NNODE * HD * 4);
  unsigned short* xb    = (unsigned short*)alloc((size_t)NPAD * HD * 2);
  unsigned short* h1b   = (unsigned short*)alloc((size_t)NPAD * HD * 2);
  int*   cnt   = (int*)alloc((size_t)NREL * NNODE * 4);
  int*   rc    = (int*)alloc(64);
  int*   poff  = (int*)alloc(64);
  int*   fc    = (int*)alloc(64);
  int*   cgr   = (int*)alloc(NG * NREL * 4);
  int*   ssrc  = (int*)alloc((size_t)EPADC * 4);
  int*   sdst  = (int*)alloc((size_t)EPADC * 4);
  float* sw    = (float*)alloc((size_t)EPADC * 4);
  int*   blockrel = (int*)alloc((size_t)NBLKE * 4);
  float* gsum  = (float*)alloc(NG * HD * 4);
  float* gmean = (float*)alloc(NG * HD * 4);
  float* relproj = (float*)alloc(NREL * LD * 4);
  float* Tf    = (float*)alloc(NREL * LD * 4);
  float* svv   = (float*)alloc(64);
  float* av    = (float*)alloc(64);
  float* PW    = (float*)alloc(NG * NREL * 4);
  float* combA = (float*)alloc((size_t)NG * 2 * LD * 4);
  float* combF = (float*)alloc((size_t)NG * OUTD * 4);
  unsigned short* Winit_t  = (unsigned short*)alloc((size_t)HD * LD * 2);
  unsigned short* Wroot1_t = (unsigned short*)alloc((size_t)LD * HD * 2);
  unsigned short* Wrel1_t  = (unsigned short*)alloc((size_t)NREL * LD * HD * 2);
  unsigned short* Wroot2_t = (unsigned short*)alloc((size_t)HD * HD * 2);
  unsigned short* Wrel2_t  = (unsigned short*)alloc((size_t)NREL * HD * HD * 2);
  float*          out2F = (float*)hb;      // alias: hb dead after layer 1
  unsigned short* As2   = As;              // alias: layer-1 gather dead after layer-1 GEMMs

  dim3 tb(256);
  // ---- conversions / transposes ----
  k_conv_bf16<<<g1d((long)NNODE * HD), tb, 0, stream>>>(x, xb, (long)NNODE * HD);
  k_transpose_bf16<<<g1d((long)HD * LD), tb, 0, stream>>>(W_init, Winit_t, 1, HD, LD);
  k_transpose_bf16<<<g1d((long)LD * HD), tb, 0, stream>>>(W_root1, Wroot1_t, 1, LD, HD);
  k_transpose_bf16<<<g1d((long)NREL * LD * HD), tb, 0, stream>>>(W_rel1, Wrel1_t, NREL, LD, HD);
  k_transpose_bf16<<<g1d((long)HD * HD), tb, 0, stream>>>(W_root2, Wroot2_t, 1, HD, HD);
  k_transpose_bf16<<<g1d((long)NREL * HD * HD), tb, 0, stream>>>(W_rel2, Wrel2_t, NREL, HD, HD);
  // ---- edge bookkeeping ----
  k_zero_i32<<<g1d((long)NREL * NNODE), tb, 0, stream>>>(cnt, (long)NREL * NNODE);
  k_zero_i32<<<1, 64, 0, stream>>>(rc, NREL);
  k_zero_i32<<<g1d(NG * NREL), tb, 0, stream>>>(cgr, NG * NREL);
  k_zero_f32<<<g1d(NG * HD), tb, 0, stream>>>(gsum, NG * HD);
  k_count<<<g1d(NEDGE), tb, 0, stream>>>(esrc, edst, eattr, batch, cnt, rc, cgr);
  k_offsets<<<1, 1, 0, stream>>>(rc, poff, fc);
  k_blockrel<<<g1d(NBLKE), tb, 0, stream>>>(poff, blockrel);
  k_init_pad<<<g1d(EPADC), tb, 0, stream>>>(ssrc, sdst, sw);
  k_build<<<g1d(NEDGE), tb, 0, stream>>>(esrc, edst, eattr, poff, fc, cnt, ssrc, sdst, sw);
  // ---- h = x @ W_init + b_init (bf16 out) ----
  dim3 wb(32, 4);
  k_gemm_wmma<<<dim3(NPAD / 32, LD / 256), wb, 0, stream>>>(
      xb, Winit_t, (float*)nullptr, hb, b_init, NNODE, HD, LD, 0);
  // ---- layer 1 ----
  k_gather<<<g1d((long)EPADC * (LD / 8)), tb, 0, stream>>>(ssrc, hb, As, LD);
  k_gemm_wmma<<<dim3(NPAD / 32, 1), wb, 0, stream>>>(
      hb, Wroot1_t, out1F, (unsigned short*)nullptr, b1, NNODE, LD, HD, 0);
  k_gemm_edges<<<dim3(NBLKE, 1), wb, 0, stream>>>(As, Wrel1_t, blockrel, sdst, sw, out1F, LD, HD);
  k_relu_bf<<<g1d((long)NNODE * HD), tb, 0, stream>>>(out1F, h1b, (long)NNODE * HD);
  // ---- layer 2 ----
  k_gather<<<g1d((long)EPADC * (HD / 8)), tb, 0, stream>>>(ssrc, h1b, As2, HD);
  k_gemm_wmma<<<dim3(NPAD / 32, 1), wb, 0, stream>>>(
      h1b, Wroot2_t, out2F, (unsigned short*)nullptr, b2, NNODE, HD, HD, 0);
  k_gemm_edges<<<dim3(NBLKE, 1), wb, 0, stream>>>(As2, Wrel2_t, blockrel, sdst, sw, out2F, HD, HD);
  k_final_h<<<g1d((long)NNODE * HD), tb, 0, stream>>>(out1F, out2F, out_nodes, (long)NNODE * HD);
  // ---- pooling + head (scalar f32 tails) ----
  k_gsum<<<g1d((long)((NNODE + GSCH - 1) / GSCH) * HD), tb, 0, stream>>>(out_nodes, batch, gsum);
  k_gmean<<<g1d(NG * HD), tb, 0, stream>>>(gsum, gptr, gmean);
  k_mm_scalar<<<g1d((long)NG * LD), tb, 0, stream>>>(gmean, Wg, bg, combA, NG, HD, LD, 2 * LD, 0);
  k_mm_scalar<<<g1d((long)NREL * LD), tb, 0, stream>>>(rel_table, W_init, b_init, relproj,
                                                       NREL, HD, LD, LD, 0);
  k_mm_scalar<<<g1d((long)NREL * LD), tb, 0, stream>>>(relproj, Wa1, ba1, Tf, NREL, LD, LD, LD, 1);
  k_mm_scalar<<<1, 64, 0, stream>>>(Tf, Wa2, ba2, svv, NREL, LD, 1, 1, 0);
  k_softmax_global<<<1, 1, 0, stream>>>(svv, rc, av);
  k_pw<<<1, 64, 0, stream>>>(cgr, av, PW);
  k_relctx<<<g1d((long)NG * LD), tb, 0, stream>>>(PW, cgr, relproj, combA);
  k_mm_scalar<<<g1d((long)NG * OUTD), tb, 0, stream>>>(combA, Wo, bo, combF,
                                                       NG, 2 * LD, OUTD, OUTD, 0);
  k_layernorm<<<NG, 256, 0, stream>>>(combF, ln_g, ln_b, out_graph);
}
